// Multi_Head_Attention_23914377904458
// MI455X (gfx1250) — matrix operand order repaired and checked by reading, not compiled, not run on hardware
//
#include <hip/hip_runtime.h>

#define B_ 4
#define S_ 2048
#define D_ 512
#define H_ 8
#define SCALE_ 0.04419417382415922f   // 1/sqrt(512)
#define NEG_   1.0e10f

typedef __attribute__((ext_vector_type(16))) __bf16 v16bf;
typedef __attribute__((ext_vector_type(8)))  float  v8f;
typedef __attribute__((ext_vector_type(8)))  unsigned short u16x8;

__device__ __forceinline__ __bf16 us2bf(unsigned short u) {
  union { unsigned short u; __bf16 h; } x; x.u = u; return x.h;
}
__device__ __forceinline__ unsigned short f2bf(float f) {
  union { float f; unsigned u; } x; x.f = f;
  unsigned r = x.u + 0x7FFFu + ((x.u >> 16) & 1u);
  return (unsigned short)(r >> 16);
}

// A-matrix (16xK slab, 16-bit): lane L holds row M=L%16.
// Halves 0..7  -> k = kk + base + 0..7   (base = 8 for lanes 16..31)
// Halves 8..15 -> k = kk + base + 16..23
__device__ __forceinline__ v16bf load_a(const unsigned short* __restrict__ row,
                                        int kk, int hi) {
  const int base = kk + (hi ? 8 : 0);
  u16x8 lo = *(const u16x8*)(row + base);
  u16x8 up = *(const u16x8*)(row + base + 16);
  v16bf a;
#pragma unroll
  for (int i = 0; i < 8; i++) { a[i] = us2bf(lo[i]); a[8 + i] = us2bf(up[i]); }
  return a;
}

// B-matrix (Kx16 slab, 16-bit), fed from a transposed ("row = output column n,
// contiguous along k") buffer: lane L holds column N=L%16.
// Halves 0..7 -> k = kk + 8*(L>=16) + 0..7; halves 8..15 -> k = kk + 8*(L>=16) + 16..23
__device__ __forceinline__ v16bf load_b(const unsigned short* __restrict__ row,
                                        int kk, int hi) {
  const int base = kk + (hi ? 8 : 0);
  u16x8 lo = *(const u16x8*)(row + base);
  u16x8 up = *(const u16x8*)(row + base + 16);
  v16bf b;
#pragma unroll
  for (int i = 0; i < 8; i++) { b[i] = us2bf(lo[i]); b[8 + i] = us2bf(up[i]); }
  return b;
}

#define WMMA_BF16(A, Bm, C) \
  __builtin_amdgcn_wmma_f32_16x16x32_bf16(false, (A), false, (Bm), (short)0, (C), false, false)

// 32x64 register-blocked GEMM tile: 2 A-fragments x 4 B-fragments -> 8 WMMAs
// per 32-deep k-step. 12 b128 loads per 8 WMMAs (~21 flop/byte at L2).
__device__ __forceinline__ void gemm_32x64(const unsigned short* __restrict__ abase,
                                           const unsigned short* __restrict__ bbase,
                                           int lda, int ldb, int K,
                                           int row0, int col0, int l16, int hi,
                                           v8f c[2][4]) {
  const unsigned short* ar0 = abase + (size_t)(row0 + l16) * lda;
  const unsigned short* ar1 = abase + (size_t)(row0 + 16 + l16) * lda;
  const unsigned short* br0 = bbase + (size_t)(col0 + l16) * ldb;
  const unsigned short* br1 = bbase + (size_t)(col0 + 16 + l16) * ldb;
  const unsigned short* br2 = bbase + (size_t)(col0 + 32 + l16) * ldb;
  const unsigned short* br3 = bbase + (size_t)(col0 + 48 + l16) * ldb;
#pragma unroll 2
  for (int kk = 0; kk < K; kk += 32) {
    v16bf a0 = load_a(ar0, kk, hi);
    v16bf a1 = load_a(ar1, kk, hi);
    v16bf b0 = load_b(br0, kk, hi);
    v16bf b1 = load_b(br1, kk, hi);
    v16bf b2 = load_b(br2, kk, hi);
    v16bf b3 = load_b(br3, kk, hi);
    c[0][0] = WMMA_BF16(a0, b0, c[0][0]);
    c[0][1] = WMMA_BF16(a0, b1, c[0][1]);
    c[0][2] = WMMA_BF16(a0, b2, c[0][2]);
    c[0][3] = WMMA_BF16(a0, b3, c[0][3]);
    c[1][0] = WMMA_BF16(a1, b0, c[1][0]);
    c[1][1] = WMMA_BF16(a1, b1, c[1][1]);
    c[1][2] = WMMA_BF16(a1, b2, c[1][2]);
    c[1][3] = WMMA_BF16(a1, b3, c[1][3]);
  }
}

__global__ void zero_kernel(float* __restrict__ o, int n) {
  int i = blockIdx.x * blockDim.x + threadIdx.x;
  if (i < n) o[i] = 0.0f;
}

__global__ void cast_x_kernel(const float* __restrict__ x,
                              unsigned short* __restrict__ xb, int n) {
  int i = blockIdx.x * blockDim.x + threadIdx.x;
  if (i < n) xb[i] = f2bf(x[i]);
}

// W[h][d][k] (f32) -> Wt[h][k][d] (bf16), so B-operand loads are contiguous.
__global__ void cast_wT_kernel(const float* __restrict__ w,
                               unsigned short* __restrict__ wt) {
  int i = blockIdx.x * blockDim.x + threadIdx.x;
  if (i < H_ * D_ * D_) {
    int h = i / (D_ * D_);
    int rem = i % (D_ * D_);
    int d = rem / D_;
    int kcol = rem % D_;
    wt[(size_t)h * D_ * D_ + (size_t)kcol * D_ + d] = f2bf(w[i]);
  }
}

// Q/K/V projection for one (b,h): C(2048x512) = xb(2048x512) @ W(512x512).
// blockIdx.y selects Q (0), K (1), V (2). V is stored transposed: Vt[k][t].
// 32x64 tile per wave: (S/32)*(D/64) = 64*8 = 512 waves -> 64 blocks.
__global__ void proj_kernel(const unsigned short* __restrict__ xb,
                            const unsigned short* __restrict__ wq,
                            const unsigned short* __restrict__ wk,
                            const unsigned short* __restrict__ wv,
                            unsigned short* __restrict__ q,
                            unsigned short* __restrict__ k,
                            unsigned short* __restrict__ vt) {
  const int which = blockIdx.y;
  const unsigned short* w = (which == 0) ? wq : ((which == 1) ? wk : wv);
  const int wave = threadIdx.x >> 5;
  const int lane = threadIdx.x & 31;
  const int tile = blockIdx.x * 8 + wave;
  const int ti = tile >> 3;                 // S/32 = 64 row tiles
  const int tj = tile & 7;                  // D/64 = 8 col tiles
  const int row0 = ti * 32, col0 = tj * 64;
  const int l16 = lane & 15, hi = lane >> 4;

  v8f c[2][4] = {};
  gemm_32x64(xb, w, D_, D_, D_, row0, col0, l16, hi, c);

  if (which == 2) {
#pragma unroll
    for (int i = 0; i < 2; i++)
#pragma unroll
      for (int j = 0; j < 4; j++)
#pragma unroll
        for (int r = 0; r < 8; r++) {
          int m = row0 + 16 * i + r + 8 * hi;
          int n = col0 + 16 * j + l16;
          vt[(size_t)n * S_ + m] = f2bf(c[i][j][r]);   // transposed store
        }
  } else {
    unsigned short* o = (which == 0) ? q : k;
#pragma unroll
    for (int i = 0; i < 2; i++)
#pragma unroll
      for (int j = 0; j < 4; j++)
#pragma unroll
        for (int r = 0; r < 8; r++) {
          int m = row0 + 16 * i + r + 8 * hi;
          int n = col0 + 16 * j + l16;
          o[(size_t)m * D_ + n] = f2bf(c[i][j][r]);
        }
  }
}

// scores(2048x2048) = Q @ K^T * scale with stride-sparse mask applied.
// B-operand of Q·Kᵀ in transposed-row form is exactly row-major K.
// 32x64 tile per wave: 64*32 = 2048 waves -> 256 blocks.
__global__ void scores_kernel(const unsigned short* __restrict__ q,
                              const unsigned short* __restrict__ kmat,
                              float* __restrict__ sc) {
  const int wave = threadIdx.x >> 5;
  const int lane = threadIdx.x & 31;
  const int tile = blockIdx.x * 8 + wave;
  const int ti = tile >> 5;                 // S/32 = 64 row tiles
  const int tj = tile & 31;                 // S/64 = 32 col tiles
  const int row0 = ti * 32, col0 = tj * 64;
  const int l16 = lane & 15, hi = lane >> 4;

  v8f c[2][4] = {};
  gemm_32x64(q, kmat, D_, D_, D_, row0, col0, l16, hi, c);

#pragma unroll
  for (int i = 0; i < 2; i++)
#pragma unroll
    for (int j = 0; j < 4; j++)
#pragma unroll
      for (int r = 0; r < 8; r++) {
        int s = row0 + 16 * i + r + 8 * hi;
        int t = col0 + 16 * j + l16;
        int d = t - s; d = (d < 0) ? -d : d;
        bool keep = (d <= 2) || ((d & 1) == 0);    // LOCAL_RANGE=2, STRIDE=2
        sc[(size_t)s * S_ + t] = keep ? c[i][j][r] * SCALE_ : -NEG_;
      }
}

// Row softmax over 2048 f32 scores -> bf16 probs. One 256-thread block/row.
__global__ void softmax_rows(const float* __restrict__ sc,
                             unsigned short* __restrict__ p) {
  __shared__ float red[256];
  const int row = blockIdx.x;
  const int tid = threadIdx.x;
  const float* srow = sc + (size_t)row * S_;
  float v[8];
  float m = -3.4e38f;
#pragma unroll
  for (int i = 0; i < 8; i++) { v[i] = srow[tid + i * 256]; m = fmaxf(m, v[i]); }
  red[tid] = m; __syncthreads();
  for (int s = 128; s > 0; s >>= 1) {
    if (tid < s) red[tid] = fmaxf(red[tid], red[tid + s]);
    __syncthreads();
  }
  m = red[0]; __syncthreads();
  float sum = 0.0f;
#pragma unroll
  for (int i = 0; i < 8; i++) { v[i] = __expf(v[i] - m); sum += v[i]; }
  red[tid] = sum; __syncthreads();
  for (int s = 128; s > 0; s >>= 1) {
    if (tid < s) red[tid] += red[tid + s];
    __syncthreads();
  }
  float inv = 1.0f / red[0];
  unsigned short* prow = p + (size_t)row * S_;
#pragma unroll
  for (int i = 0; i < 8; i++) prow[tid + i * 256] = f2bf(v[i] * inv);
}

// out(2048x512) += (probs @ V) / H.  Vt[k][t] gives contiguous B loads.
// Each out element is touched once per launch; launches serialize on the
// stream, so plain read-modify-write is safe. 512 waves -> 64 blocks.
__global__ void av_kernel(const unsigned short* __restrict__ p,
                          const unsigned short* __restrict__ vt,
                          float* __restrict__ outb) {
  const int wave = threadIdx.x >> 5;
  const int lane = threadIdx.x & 31;
  const int tile = blockIdx.x * 8 + wave;
  const int ti = tile >> 3;                 // S/32 = 64 row tiles
  const int tj = tile & 7;                  // D/64 = 8 col tiles
  const int row0 = ti * 32, col0 = tj * 64;
  const int l16 = lane & 15, hi = lane >> 4;

  v8f c[2][4] = {};
  gemm_32x64(p, vt, S_, S_, S_, row0, col0, l16, hi, c);

  const float invH = 1.0f / (float)H_;
#pragma unroll
  for (int i = 0; i < 2; i++)
#pragma unroll
    for (int j = 0; j < 4; j++)
#pragma unroll
      for (int r = 0; r < 8; r++) {
        int s = row0 + 16 * i + r + 8 * hi;
        int dcol = col0 + 16 * j + l16;
        outb[(size_t)s * D_ + dcol] += c[i][j][r] * invH;
      }
}

extern "C" void kernel_launch(void* const* d_in, const int* in_sizes, int n_in,
                              void* d_out, int out_size, void* d_ws, size_t ws_size,
                              hipStream_t stream) {
  (void)in_sizes; (void)n_in; (void)ws_size;
  const float* x  = (const float*)d_in[0];
  const float* wq = (const float*)d_in[1];
  const float* wk = (const float*)d_in[2];
  const float* wv = (const float*)d_in[3];
  float* out = (float*)d_out;

  char* ws = (char*)d_ws;
  size_t o = 0;
  unsigned short* xb  = (unsigned short*)(ws + o); o += (size_t)B_ * S_ * D_ * 2;
  unsigned short* wqt = (unsigned short*)(ws + o); o += (size_t)H_ * D_ * D_ * 2;
  unsigned short* wkt = (unsigned short*)(ws + o); o += (size_t)H_ * D_ * D_ * 2;
  unsigned short* wvt = (unsigned short*)(ws + o); o += (size_t)H_ * D_ * D_ * 2;
  unsigned short* qb  = (unsigned short*)(ws + o); o += (size_t)S_ * D_ * 2;
  unsigned short* kb  = (unsigned short*)(ws + o); o += (size_t)S_ * D_ * 2;
  unsigned short* vtb = (unsigned short*)(ws + o); o += (size_t)S_ * D_ * 2;
  float*          sc  = (float*)(ws + o);         o += (size_t)S_ * S_ * 4;
  unsigned short* pb  = (unsigned short*)(ws + o); o += (size_t)S_ * S_ * 2;
  // total workspace: ~52.4 MB

  zero_kernel<<<(out_size + 255) / 256, 256, 0, stream>>>(out, out_size);

  const int nx = B_ * S_ * D_;
  cast_x_kernel<<<(nx + 255) / 256, 256, 0, stream>>>(x, xb, nx);
  const int nw = H_ * D_ * D_;
  cast_wT_kernel<<<(nw + 255) / 256, 256, 0, stream>>>(wq, wqt);
  cast_wT_kernel<<<(nw + 255) / 256, 256, 0, stream>>>(wk, wkt);
  cast_wT_kernel<<<(nw + 255) / 256, 256, 0, stream>>>(wv, wvt);

  for (int b = 0; b < B_; b++) {
    for (int h = 0; h < H_; h++) {
      proj_kernel<<<dim3(64, 3, 1), 256, 0, stream>>>(
          xb + (size_t)b * S_ * D_,
          wqt + (size_t)h * D_ * D_,
          wkt + (size_t)h * D_ * D_,
          wvt + (size_t)h * D_ * D_,
          qb, kb, vtb);
      scores_kernel<<<256, 256, 0, stream>>>(qb, kb, sc);
      softmax_rows<<<2048, 256, 0, stream>>>(sc, pb);
      av_kernel<<<64, 256, 0, stream>>>(pb, vtb, out + (size_t)b * S_ * D_);
    }
  }
}
